// SpatialGNN_84653805404492
// MI455X (gfx1250) — matrix operand order repaired and checked by reading, not compiled, not run on hardware
//
#include <hip/hip_runtime.h>
#include <hip/hip_bf16.h>
#include <math.h>

// ---------------------------------------------------------------------------
// CDNA5 WMMA types
// ---------------------------------------------------------------------------
typedef __attribute__((ext_vector_type(8)))  __bf16 v8bf;
typedef __attribute__((ext_vector_type(16))) __bf16 v16bf;
typedef __attribute__((ext_vector_type(8)))  float  v8f;

#define BM 128
#define BN 128
#define BK 32
#define LDR 40          // padded LDS row stride in bf16 elements (80B; 16B-aligned groups)
#define BN_EPS 1e-5f

// ---------------------------------------------------------------------------
// GEMM: H[M,N] = X[M,K] @ W[K,N]  (W pre-transposed+bf16: Wt[N][K])
// 256 threads = 8 wave32s; 128x128 block tile; wave = 32x64 patch =
// 2x4 v_wmma_f32_16x16x32_bf16 accumulators. Double-buffered LDS.
// CONV=1: X fp32 (layer 0), register staging + on-the-fly bf16 convert,
//         handles K%32 != 0 tail.
// CONV=0: X bf16 (layers 1-3, K%32==0), staging via CDNA5 async
//         GLOBAL_LOAD_ASYNC_TO_LDS_B128 (ASYNCcnt), no VGPR round trip.
// ---------------------------------------------------------------------------
template <int CONV>
__global__ __launch_bounds__(256)
void gcn_gemm_wmma(const void* __restrict__ Xv, const __bf16* __restrict__ Wt,
                   float* __restrict__ H, int M, int K, int N) {
  __shared__ __align__(16) __bf16 As[2][BM * LDR];   // As[m][k]
  __shared__ __align__(16) __bf16 Bs[2][BN * LDR];   // Bs[n][k]

  const int t    = threadIdx.x;
  const int wave = t >> 5;
  const int lane = t & 31;
  const int lm   = lane & 15;
  const int lhi  = lane >> 4;

  const int tileN = blockIdx.x * BN;   // N-tiles fastest => A panel L2 reuse
  const int tileM = blockIdx.y * BM;
  const int wm = (wave >> 1) * 32;     // 0,32,64,96
  const int wn = (wave & 1) * 64;      // 0,64

  const float*  Xf = (const float*)Xv;
  const __bf16* Xb = (const __bf16*)Xv;

  // staging chunks: c = t, t+256 -> row = c>>2 (0..127), koff = (c&3)*8
  const int row0 = t >> 2;
  const int row1 = row0 + 64;
  const int koff = (t & 3) * 8;
  const bool mfull = (tileM + BM <= M);

  v8f acc[2][4] = {};

  float4 afA[2][2];           // CONV staging regs (fp32)
  v8bf   bRc[2];              // CONV staging regs for B

  // ---- register-staging path (CONV=1) ----
  auto loadStage = [&](int k0) {
    const int rows[2] = {row0, row1};
    const bool kok = (k0 + koff + 8 <= K);
    for (int i = 0; i < 2; i++) {
      const int gr = tileM + rows[i];
      const bool aok = kok && (mfull || gr < M);
      if (aok) {
        const float* p = Xf + (size_t)gr * K + k0 + koff;
        afA[i][0] = *(const float4*)(p);
        afA[i][1] = *(const float4*)(p + 4);
      } else {
        afA[i][0] = make_float4(0.f, 0.f, 0.f, 0.f);
        afA[i][1] = make_float4(0.f, 0.f, 0.f, 0.f);
      }
      const int gn = tileN + rows[i];
      v8bf zb = {};
      bRc[i] = (kok && gn < N) ? *(const v8bf*)(Wt + (size_t)gn * K + k0 + koff) : zb;
    }
  };

  auto storeStage = [&](int buf) {
    const int rows[2] = {row0, row1};
    for (int i = 0; i < 2; i++) {
      __bf16* pa = &As[buf][rows[i] * LDR + koff];
      v8bf r;
      r[0] = (__bf16)afA[i][0].x; r[1] = (__bf16)afA[i][0].y;
      r[2] = (__bf16)afA[i][0].z; r[3] = (__bf16)afA[i][0].w;
      r[4] = (__bf16)afA[i][1].x; r[5] = (__bf16)afA[i][1].y;
      r[6] = (__bf16)afA[i][1].z; r[7] = (__bf16)afA[i][1].w;
      *(v8bf*)pa = r;
      *(v8bf*)(&Bs[buf][rows[i] * LDR + koff]) = bRc[i];
    }
  };

  // ---- async-staging path (CONV=0): global -> LDS without VGPR round trip ----
  // Each lane async-copies one 16B chunk; guarded-off chunks are skipped
  // (stale LDS only feeds epilogue-guarded rows/cols; K%32==0 so no K-tail).
  auto asyncStage = [&](int k0, int buf) {
    const int rows[2] = {row0, row1};
    for (int i = 0; i < 2; i++) {
      const int gr = tileM + rows[i];
      if (mfull || gr < M) {
        unsigned ldsa = (unsigned)(uintptr_t)&As[buf][rows[i] * LDR + koff];
        const __bf16* gp = Xb + (size_t)gr * K + k0 + koff;
        asm volatile("global_load_async_to_lds_b128 %0, %1, off"
                     :: "v"(ldsa), "v"(gp) : "memory");
      }
      const int gn = tileN + rows[i];
      if (gn < N) {
        unsigned ldsb = (unsigned)(uintptr_t)&Bs[buf][rows[i] * LDR + koff];
        const __bf16* gp = Wt + (size_t)gn * K + k0 + koff;
        asm volatile("global_load_async_to_lds_b128 %0, %1, off"
                     :: "v"(ldsb), "v"(gp) : "memory");
      }
    }
  };

  auto compute = [&](int buf) {
    // A 16x32 bf16 layout: lane holds M=l&15; elems 0..7=K[hi*8..], 8..15=K[16+hi*8..]
    v16bf a[2];
    for (int i = 0; i < 2; i++) {
      const __bf16* base = &As[buf][(wm + i * 16 + lm) * LDR + lhi * 8];
      v8bf x0 = *(const v8bf*)(base);
      v8bf x1 = *(const v8bf*)(base + 16);
      a[i] = __builtin_shufflevector(x0, x1, 0,1,2,3,4,5,6,7,8,9,10,11,12,13,14,15);
    }
    // B 32x16 bf16 layout: lane holds N=l&15; elems 0..7=K[hi*8..], 8..15=K[16+hi*8..]
    v16bf b[4];
    for (int j = 0; j < 4; j++) {
      const __bf16* base = &Bs[buf][(wn + j * 16 + lm) * LDR + lhi * 8];
      v8bf x0 = *(const v8bf*)(base);
      v8bf x1 = *(const v8bf*)(base + 16);
      b[j] = __builtin_shufflevector(x0, x1, 0,1,2,3,4,5,6,7,8,9,10,11,12,13,14,15);
    }
    for (int i = 0; i < 2; i++)
      for (int j = 0; j < 4; j++)
        acc[i][j] = __builtin_amdgcn_wmma_f32_16x16x32_bf16(
            false, a[i], false, b[j], (short)0, acc[i][j], false, false);
  };

  const int nk = (K + BK - 1) >> 5;

  if constexpr (CONV) {
    loadStage(0);
    storeStage(0);
    __syncthreads();
    for (int kt = 0; kt < nk; kt++) {
      const int cur = kt & 1;
      const bool nxt = (kt + 1 < nk);
      if (nxt) loadStage((kt + 1) << 5);   // global b128 loads in flight...
      compute(cur);                         // ...while WMMAs execute
      if (nxt) storeStage(cur ^ 1);
      __syncthreads();
    }
  } else {
    asyncStage(0, 0);
    asm volatile("s_wait_asynccnt 0" ::: "memory");
    __syncthreads();
    for (int kt = 0; kt < nk; kt++) {
      const int cur = kt & 1;
      if (kt + 1 < nk) asyncStage((kt + 1) << 5, cur ^ 1);  // async DMA next tile
      compute(cur);                                          // WMMAs overlap DMA
      asm volatile("s_wait_asynccnt 0" ::: "memory");
      __syncthreads();
    }
  }

  // C layout: VGPR r -> M = base + hi*8 + r, N = base + (lane&15)
  for (int i = 0; i < 2; i++) {
    const int mb = tileM + wm + i * 16 + lhi * 8;
    for (int j = 0; j < 4; j++) {
      const int n = tileN + wn + j * 16 + lm;
      if (n < N) {
        for (int r = 0; r < 8; r++) {
          const int m = mb + r;
          if (m < M) H[(size_t)m * N + n] = acc[i][j][r];
        }
      }
    }
  }
}

// ---------------------------------------------------------------------------
// W[K,N] fp32 -> Wt[N,K] bf16 (tiny, once per layer)
// ---------------------------------------------------------------------------
__global__ void wtrans_kernel(const float* __restrict__ W, __bf16* __restrict__ Wt,
                              int K, int N, int total) {
  int idx = blockIdx.x * blockDim.x + threadIdx.x;
  if (idx >= total) return;
  int n = idx / K;
  int k = idx - n * K;
  Wt[idx] = (__bf16)W[(size_t)k * N + n];
}

// ---------------------------------------------------------------------------
// Graph / elementwise kernels
// ---------------------------------------------------------------------------
__global__ void init_deg_kernel(float* deg, int Nn) {
  int i = blockIdx.x * blockDim.x + threadIdx.x;
  if (i < Nn) deg[i] = 1.0f;   // self loop
}

__global__ void deg_accum_kernel(float* deg, const int* __restrict__ dst, int E) {
  int e = blockIdx.x * blockDim.x + threadIdx.x;
  if (e < E) atomicAdd(&deg[dst[e]], 1.0f);
}

__global__ void dinv_kernel(float* dinv, int Nn) {
  int i = blockIdx.x * blockDim.x + threadIdx.x;
  if (i < Nn) dinv[i] = rsqrtf(dinv[i]);
}

__global__ void coef_kernel(float* __restrict__ coef, const float* __restrict__ dinv,
                            const int* __restrict__ src, const int* __restrict__ dst, int E) {
  int e = blockIdx.x * blockDim.x + threadIdx.x;
  if (e < E) coef[e] = dinv[src[e]] * dinv[dst[e]];
}

// A[i, f4] = H[i, f4] * dinv[i]^2  (float4; also initializes A)
__global__ void selfinit_kernel(const float* __restrict__ H, const float* __restrict__ dinv,
                                float* __restrict__ A, int total4, int logFq) {
  int idx = blockIdx.x * blockDim.x + threadIdx.x;
  if (idx >= total4) return;
  int i = idx >> logFq;
  float d = dinv[i];
  float d2 = d * d;
  float4 h = ((const float4*)H)[idx];
  float4 o;
  o.x = h.x * d2; o.y = h.y * d2; o.z = h.z * d2; o.w = h.w * d2;
  ((float4*)A)[idx] = o;
}

// A[dst, f4] += H[src, f4] * coef[e]   (4 features / thread, f32 atomics)
__global__ void scatter_kernel(const float* __restrict__ H, float* __restrict__ A,
                               const int* __restrict__ src, const int* __restrict__ dst,
                               const float* __restrict__ coef,
                               int total4, int Fq, int logFq, int F) {
  int idx = blockIdx.x * blockDim.x + threadIdx.x;
  if (idx >= total4) return;
  int e  = idx >> logFq;
  int f4 = (idx & (Fq - 1)) << 2;
  float c = coef[e];
  const float* hp = H + (size_t)src[e] * F + f4;
  float4 h = *(const float4*)hp;
  float* ap = A + (size_t)dst[e] * F + f4;
  atomicAdd(ap + 0, h.x * c);
  atomicAdd(ap + 1, h.y * c);
  atomicAdd(ap + 2, h.z * c);
  atomicAdd(ap + 3, h.w * c);
}

// val = BN(A + bias)[+ReLU]; write bf16 (next GEMM input) or fp32 in place
__global__ void post_kernel(float* __restrict__ A, __bf16* __restrict__ Ob,
                            const float* __restrict__ b, const float* __restrict__ g,
                            const float* __restrict__ be, const float* __restrict__ m,
                            const float* __restrict__ v,
                            int total, int F, int logF, int doRelu, int toBf) {
  int idx = blockIdx.x * blockDim.x + threadIdx.x;
  if (idx >= total) return;
  int f = idx & (F - 1);
  float val = A[idx] + b[f];
  val = (val - m[f]) * rsqrtf(v[f] + BN_EPS) * g[f] + be[f];
  if (doRelu) val = fmaxf(val, 0.f);
  if (toBf) Ob[idx] = (__bf16)val;
  else      A[idx]  = val;
}

__global__ void zero_emb_kernel(float* emb) {
  if (threadIdx.x < 64) emb[threadIdx.x] = 0.f;
}

__global__ void pool_kernel(const float* __restrict__ Hin, float* __restrict__ emb, int Nn) {
  __shared__ float sred[256];
  int t = threadIdx.x;
  int f = t & 63;
  int seg = t >> 6;
  float s = 0.f;
  for (int r = blockIdx.x * 4 + seg; r < Nn; r += gridDim.x * 4)
    s += Hin[(size_t)r * 64 + f];
  sred[t] = s;
  __syncthreads();
  if (seg == 0) {
    float tot = sred[f] + sred[64 + f] + sred[128 + f] + sred[192 + f];
    atomicAdd(&emb[f], tot);
  }
}

__global__ void head_kernel(const float* __restrict__ emb,
                            const float* __restrict__ Wc1, const float* __restrict__ bc1,
                            const float* __restrict__ Wc2, const float* __restrict__ bc2,
                            const float* __restrict__ Wr1, const float* __restrict__ br1,
                            const float* __restrict__ Wr2, const float* __restrict__ br2,
                            float* __restrict__ out, float invN) {
  __shared__ float se[64], hc[32], hr[32];
  int t = threadIdx.x;
  if (t < 64) se[t] = emb[t] * invN;
  __syncthreads();
  if (t < 32) {
    float sc = bc1[t], sr = br1[t];
    for (int k = 0; k < 64; k++) {
      float e = se[k];
      sc += e * Wc1[k * 32 + t];
      sr += e * Wr1[k * 32 + t];
    }
    hc[t] = fmaxf(sc, 0.f);
    hr[t] = fmaxf(sr, 0.f);
  }
  __syncthreads();
  if (t < 10) {
    float s = bc2[t];
    for (int j = 0; j < 32; j++) s += hc[j] * Wc2[j * 10 + t];
    out[t] = s;
  }
  if (t == 32) {
    float s = br2[0];
    for (int j = 0; j < 32; j++) s += hr[j] * Wr2[j];
    out[10] = 1.f / (1.f + expf(-s));
  }
}

// ---------------------------------------------------------------------------
// Orchestration
// ---------------------------------------------------------------------------
extern "C" void kernel_launch(void* const* d_in, const int* in_sizes, int n_in,
                              void* d_out, int out_size, void* d_ws, size_t ws_size,
                              hipStream_t stream) {
  const float* x  = (const float*)d_in[0];
  const int*   ei = (const int*)d_in[1];      // [2,E] int32: row0=src, row1=dst
  const int Nn = in_sizes[0] / 2000;          // 50000
  const int E  = in_sizes[1] / 2;             // 400000
  const int* src = ei;
  const int* dst = ei + E;

  const float* Ws[4]  = {(const float*)d_in[2],  (const float*)d_in[8],
                         (const float*)d_in[14], (const float*)d_in[20]};
  const float* bs[4]  = {(const float*)d_in[3],  (const float*)d_in[9],
                         (const float*)d_in[15], (const float*)d_in[21]};
  const float* gs[4]  = {(const float*)d_in[4],  (const float*)d_in[10],
                         (const float*)d_in[16], (const float*)d_in[22]};
  const float* bes[4] = {(const float*)d_in[5],  (const float*)d_in[11],
                         (const float*)d_in[17], (const float*)d_in[23]};
  const float* ms[4]  = {(const float*)d_in[6],  (const float*)d_in[12],
                         (const float*)d_in[18], (const float*)d_in[24]};
  const float* vs[4]  = {(const float*)d_in[7],  (const float*)d_in[13],
                         (const float*)d_in[19], (const float*)d_in[25]};
  const float* Wc1 = (const float*)d_in[26];
  const float* bc1 = (const float*)d_in[27];
  const float* Wc2 = (const float*)d_in[28];
  const float* bc2 = (const float*)d_in[29];
  const float* Wr1 = (const float*)d_in[30];
  const float* br1 = (const float*)d_in[31];
  const float* Wr2 = (const float*)d_in[32];
  const float* br2 = (const float*)d_in[33];
  float* out = (float*)d_out;

  // ---- workspace carve ----
  char* w = (char*)d_ws;
  auto carve = [&](size_t bytes) -> void* {
    void* p = (void*)w;
    w += (bytes + 255) & ~(size_t)255;
    return p;
  };
  const int dims[5] = {2000, 512, 256, 128, 64};
  float*  dinv = (float*)carve((size_t)Nn * 4);
  float*  coef = (float*)carve((size_t)E * 4);
  float*  emb  = (float*)carve(256);
  float*  Hb   = (float*)carve((size_t)Nn * 512 * 4);   // GEMM output
  float*  Afp  = (float*)carve((size_t)Nn * 512 * 4);   // aggregation (fp32)
  __bf16* Xp   = (__bf16*)carve((size_t)Nn * 512 * 2);  // bf16 activations ping
  __bf16* Xq   = (__bf16*)carve((size_t)Nn * 256 * 2);  // bf16 activations pong
  __bf16* Wt[4];
  for (int l = 0; l < 4; l++)
    Wt[l] = (__bf16*)carve((size_t)dims[l] * dims[l + 1] * 2);

  const int TB = 256;

  // ---- W transpose + bf16 convert (tiny) ----
  for (int l = 0; l < 4; l++) {
    int tot = dims[l] * dims[l + 1];
    wtrans_kernel<<<(tot + TB - 1) / TB, TB, 0, stream>>>(Ws[l], Wt[l], dims[l], dims[l + 1], tot);
  }

  // ---- degrees and norm coefficients ----
  init_deg_kernel<<<(Nn + TB - 1) / TB, TB, 0, stream>>>(dinv, Nn);
  deg_accum_kernel<<<(E + TB - 1) / TB, TB, 0, stream>>>(dinv, dst, E);
  dinv_kernel<<<(Nn + TB - 1) / TB, TB, 0, stream>>>(dinv, Nn);
  coef_kernel<<<(E + TB - 1) / TB, TB, 0, stream>>>(coef, dinv, src, dst, E);

  // ---- 4 GCN layers ----
  __bf16* XbIn[4]  = {nullptr, Xp, Xq, Xp};   // bf16 GEMM inputs (l>=1)
  __bf16* XbOut[4] = {Xp, Xq, Xp, nullptr};   // bf16 post outputs (l<=2)
  for (int l = 0; l < 4; l++) {
    const int K = dims[l];
    const int F = dims[l + 1];
    const int logF = (F == 512) ? 9 : (F == 256) ? 8 : (F == 128) ? 7 : 6;

    dim3 gg((F + BN - 1) / BN, (Nn + BM - 1) / BM);   // N-tiles fastest
    if (l == 0)
      gcn_gemm_wmma<1><<<gg, 256, 0, stream>>>((const void*)x, Wt[l], Hb, Nn, K, F);
    else
      gcn_gemm_wmma<0><<<gg, 256, 0, stream>>>((const void*)XbIn[l], Wt[l], Hb, Nn, K, F);

    const int totNF4 = (Nn * F) >> 2;
    selfinit_kernel<<<(totNF4 + TB - 1) / TB, TB, 0, stream>>>(Hb, dinv, Afp, totNF4, logF - 2);

    const int totEF4 = (E >> 2) * F;   // E*F/4, max 51.2M
    scatter_kernel<<<(totEF4 + TB - 1) / TB, TB, 0, stream>>>(Hb, Afp, src, dst, coef,
                                                              totEF4, F >> 2, logF - 2, F);

    const int totNF = Nn * F;
    post_kernel<<<(totNF + TB - 1) / TB, TB, 0, stream>>>(Afp, XbOut[l], bs[l], gs[l], bes[l],
                                                          ms[l], vs[l], totNF, F, logF,
                                                          (l < 3) ? 1 : 0, (l < 3) ? 1 : 0);
  }

  // ---- mean pool + heads ----
  zero_emb_kernel<<<1, 64, 0, stream>>>(emb);
  pool_kernel<<<256, 256, 0, stream>>>(Afp, emb, Nn);
  head_kernel<<<1, 64, 0, stream>>>(emb, Wc1, bc1, Wc2, bc2, Wr1, br1, Wr2, br2,
                                    out, 1.0f / (float)Nn);
}
